// FirstNet_1434519076932
// MI455X (gfx1250) — matrix operand order repaired and checked by reading, not compiled, not run on hardware
//
#include <hip/hip_runtime.h>
#include <math.h>

typedef __attribute__((ext_vector_type(16))) _Float16 v16h;
typedef __attribute__((ext_vector_type(8)))  float    v8f;
typedef __attribute__((ext_vector_type(4)))  float    f4;

// ---------------------------------------------------------------------------
// Utility kernels
// ---------------------------------------------------------------------------
__global__ void k_fill(float* __restrict__ p, float v, int n) {
    int t = blockIdx.x * blockDim.x + threadIdx.x;
    if (t < n) p[t] = v;
}

__global__ void k_deg(const int* __restrict__ col, float* __restrict__ deg, int E) {
    int t = blockIdx.x * blockDim.x + threadIdx.x;
    if (t < E) atomicAdd(&deg[col[t]], 1.0f);
}

__global__ void k_dinv(const float* __restrict__ deg, float* __restrict__ dinv, int n) {
    int t = blockIdx.x * blockDim.x + threadIdx.x;
    if (t < n) dinv[t] = rsqrtf(deg[t]);   // deg >= 1 always (self-loop)
}

// ---------------------------------------------------------------------------
// WMMA GEMM: H[N,OUT] = act(X[N,K]) @ W[K,OUT]
// One wave computes a 16x16 output tile. K processed in chunks of 32.
// A-fragment layout (16-bit A 16x32, ISA 7.12.2):
//   lane = half*16 + m ; element e: K = kc + e + 8*half        (e<8)
//                                   K = kc + e + 8 + 8*half    (e>=8)
// B-fragment layout: lane n=lane&15, element e: same K as A fragment
// D layout: element v -> row m = v + 8*half, col n = lane&15
// ---------------------------------------------------------------------------
template<int K, int OUT, bool RELU>
__global__ __launch_bounds__(256)
void gemm_wmma(const float* __restrict__ X, const float* __restrict__ W,
               float* __restrict__ H, int nRows)
{
    constexpr int CB = (OUT + 15) / 16;
    const int lane  = threadIdx.x & 31;
    const int wave  = blockIdx.x * (blockDim.x >> 5) + (threadIdx.x >> 5);
    const int tiles = (nRows / 16) * CB;
    if (wave >= tiles) return;                 // uniform per wave: EXEC stays full

    const int rb   = wave / CB;
    const int cb   = wave % CB;
    const int half = lane >> 4;
    const int m    = lane & 15;                // A row within tile; also B/D column
    const int row  = rb * 16 + m;

    v8f acc = {};

    for (int kc = 0; kc < K; kc += 32) {
        // ---- A fragment ----
        float av[16];
        if constexpr ((K % 32) == 0) {
            const float* p = X + (size_t)row * K + kc + 8 * half;
            f4 x0 = *(const f4*)(p);
            f4 x1 = *(const f4*)(p + 4);
            f4 x2 = *(const f4*)(p + 16);
            f4 x3 = *(const f4*)(p + 20);
            av[0]=x0.x; av[1]=x0.y; av[2]=x0.z; av[3]=x0.w;
            av[4]=x1.x; av[5]=x1.y; av[6]=x1.z; av[7]=x1.w;
            av[8]=x2.x; av[9]=x2.y; av[10]=x2.z; av[11]=x2.w;
            av[12]=x3.x; av[13]=x3.y; av[14]=x3.z; av[15]=x3.w;
        } else {
            #pragma unroll
            for (int e = 0; e < 16; ++e) {
                int k = kc + ((e < 8) ? (e + 8 * half) : (e + 8 + 8 * half));
                av[e] = (k < K) ? X[(size_t)row * K + k] : 0.0f;
            }
        }
        v16h a;
        #pragma unroll
        for (int e = 0; e < 16; ++e) {
            float v = av[e];
            if (RELU) v = fmaxf(v, 0.0f);
            a[e] = (_Float16)v;
        }

        // ---- B fragment ----
        v16h b;
        #pragma unroll
        for (int e = 0; e < 16; ++e) {
            int k = kc + ((e < 8) ? (e + 8 * half) : (e + 8 + 8 * half));
            float v = 0.0f;
            if (k < K && (cb * 16 + m) < OUT) v = W[(size_t)k * OUT + cb * 16 + m];
            b[e] = (_Float16)v;
        }

        acc = __builtin_amdgcn_wmma_f32_16x16x32_f16(
                  /*neg_a=*/false, a, /*neg_b=*/false, b,
                  /*c_mod=*/(short)0, acc, /*reuse_a=*/false, /*reuse_b=*/false);
    }

    #pragma unroll
    for (int v = 0; v < 8; ++v) {
        int mm  = v + 8 * half;
        int col = cb * 16 + m;
        if (col < OUT)
            H[(size_t)(rb * 16 + mm) * OUT + col] = acc[v];
    }
}

template<int K, int OUT, bool RELU>
static void launch_gemm(const float* X, const float* W, float* H, int N, hipStream_t s) {
    constexpr int CB = (OUT + 15) / 16;
    int tiles  = (N / 16) * CB;
    int blocks = (tiles + 7) / 8;              // 8 waves (256 threads) per block
    gemm_wmma<K, OUT, RELU><<<blocks, 256, 0, s>>>(X, W, H, N);
}

// ---------------------------------------------------------------------------
// Aggregation: agg[i] = h[i]*dinv[i]^2 + bias  (self-loop + bias, full overwrite)
// then edge scatter: agg[col] += h[row] * dinv[row]*dinv[col]
// ---------------------------------------------------------------------------
template<int OUT>
__global__ void init_agg(const float* __restrict__ h, const float* __restrict__ dinv,
                         const float* __restrict__ bias, float* __restrict__ agg, int n)
{
    int t = blockIdx.x * blockDim.x + threadIdx.x;
    if (t >= n * OUT) return;
    int i = t / OUT, f = t % OUT;
    float d = dinv[i];
    agg[t] = h[t] * d * d + bias[f];
}

template<int OUT>
__global__ void scatter_edges(const float* __restrict__ h, const int* __restrict__ row,
                              const int* __restrict__ col, const float* __restrict__ dinv,
                              float* __restrict__ agg, int E)
{
    constexpr int FP = OUT / 4;                // float4 groups per edge
    int t = blockIdx.x * blockDim.x + threadIdx.x;
    if (t >= E * FP) return;
    int e  = t / FP;
    int fo = (t % FP) * 4;
    int r = row[e], c = col[e];
    float w = dinv[r] * dinv[c];
    f4 v = *(const f4*)(h + (size_t)r * OUT + fo);
    float* dst = agg + (size_t)c * OUT + fo;
    atomicAdd(dst + 0, v.x * w);
    atomicAdd(dst + 1, v.y * w);
    atomicAdd(dst + 2, v.z * w);
    atomicAdd(dst + 3, v.w * w);
}

// ---------------------------------------------------------------------------
// Pooling (segment max) + log_softmax
// ---------------------------------------------------------------------------
__device__ inline void atomicMaxF(float* addr, float value) {
    if (value >= 0.0f)
        atomicMax((int*)addr, __float_as_int(value));
    else
        atomicMin((unsigned int*)addr, __float_as_uint(value));
}

__global__ void k_pool(const float* __restrict__ h4, const int* __restrict__ batch,
                       float* __restrict__ out, int n)
{
    int t = blockIdx.x * blockDim.x + threadIdx.x;
    if (t >= n * 4) return;
    int i = t / 4, c = t % 4;
    atomicMaxF(&out[(size_t)batch[i] * 4 + c], h4[t]);
}

__global__ void k_logsoftmax(float* __restrict__ out, int G) {
    int g = blockIdx.x * blockDim.x + threadIdx.x;
    if (g >= G) return;
    float v[4];
    #pragma unroll
    for (int c = 0; c < 4; ++c) v[c] = out[g * 4 + c];
    float mx = fmaxf(fmaxf(v[0], v[1]), fmaxf(v[2], v[3]));
    float s = 0.0f;
    #pragma unroll
    for (int c = 0; c < 4; ++c) s += expf(v[c] - mx);
    float l = logf(s);
    #pragma unroll
    for (int c = 0; c < 4; ++c) out[g * 4 + c] = v[c] - mx - l;
}

// ---------------------------------------------------------------------------
// Entry point
// ---------------------------------------------------------------------------
static inline int grid256(long long n) { return (int)((n + 255) / 256); }

extern "C" void kernel_launch(void* const* d_in, const int* in_sizes, int n_in,
                              void* d_out, int out_size, void* d_ws, size_t ws_size,
                              hipStream_t stream)
{
    const float* x    = (const float*)d_in[0];
    const int*   ei   = (const int*)d_in[1];
    const int*   batch= (const int*)d_in[2];
    const float* W1 = (const float*)d_in[3];  const float* b1 = (const float*)d_in[4];
    const float* W2 = (const float*)d_in[5];  const float* b2 = (const float*)d_in[6];
    const float* W3 = (const float*)d_in[7];  const float* b3 = (const float*)d_in[8];
    const float* W4 = (const float*)d_in[9];  const float* b4 = (const float*)d_in[10];

    const int N = in_sizes[0] / 128;          // 100000
    const int E = in_sizes[1] / 2;            // 3200000
    const int G = 128;
    const int* row = ei;                      // edge_index[0] = source
    const int* col = ei + E;                  // edge_index[1] = target

    float* ws   = (float*)d_ws;
    float* deg  = ws;
    float* dinv = ws + N;
    float* h    = ws + (size_t)2 * N;
    float* agg0 = h    + (size_t)N * 64;
    float* agg1 = agg0 + (size_t)N * 64;
    float* out  = (float*)d_out;

    // ---- GCN symmetric normalization ----
    k_fill<<<grid256(N), 256, 0, stream>>>(deg, 1.0f, N);           // self-loop
    k_deg <<<grid256(E), 256, 0, stream>>>(col, deg, E);
    k_dinv<<<grid256(N), 256, 0, stream>>>(deg, dinv, N);

    // ---- Layer 1: 128 -> 16 ----
    launch_gemm<128, 16, false>(x, W1, h, N, stream);
    init_agg<16><<<grid256((long long)N * 16), 256, 0, stream>>>(h, dinv, b1, agg0, N);
    scatter_edges<16><<<grid256((long long)E * 4), 256, 0, stream>>>(h, row, col, dinv, agg0, E);

    // ---- Layer 2: 16 -> 32 (relu of agg0 fused into A load) ----
    launch_gemm<16, 32, true>(agg0, W2, h, N, stream);
    init_agg<32><<<grid256((long long)N * 32), 256, 0, stream>>>(h, dinv, b2, agg1, N);
    scatter_edges<32><<<grid256((long long)E * 8), 256, 0, stream>>>(h, row, col, dinv, agg1, E);

    // ---- Layer 3: 32 -> 64 ----
    launch_gemm<32, 64, true>(agg1, W3, h, N, stream);
    init_agg<64><<<grid256((long long)N * 64), 256, 0, stream>>>(h, dinv, b3, agg0, N);
    scatter_edges<64><<<grid256((long long)E * 16), 256, 0, stream>>>(h, row, col, dinv, agg0, E);

    // ---- Layer 4: 64 -> 4 (no relu after) ----
    launch_gemm<64, 4, true>(agg0, W4, h, N, stream);
    init_agg<4><<<grid256((long long)N * 4), 256, 0, stream>>>(h, dinv, b4, agg1, N);
    scatter_edges<4><<<grid256((long long)E * 1), 256, 0, stream>>>(h, row, col, dinv, agg1, E);

    // ---- Global max pooling + log_softmax ----
    k_fill<<<grid256(G * 4), 256, 0, stream>>>(out, -__builtin_inff(), G * 4);
    k_pool<<<grid256((long long)N * 4), 256, 0, stream>>>(agg1, batch, out, N);
    k_logsoftmax<<<1, 128, 0, stream>>>(out, G);
}
